// BitExpert_33191507264221
// MI455X (gfx1250) — 8-bit matrix operand order repaired and checked by reading, compiled, not run on hardware
//
#include <hip/hip_runtime.h>
#include <hip/hip_bf16.h>
#include <stdint.h>


typedef __attribute__((ext_vector_type(8))) int v8i;

#define EPSQ 1e-5f
#define NSUB 4


__global__ void absmean_partial_kernel(const float* __restrict__ w,
                                       float* __restrict__ partial, size_t n) {
    __shared__ float smem[256];
    float acc = 0.0f;
    for (size_t i = (size_t)blockIdx.x * 256 + threadIdx.x; i < n;
         i += (size_t)gridDim.x * 256)
        acc += fabsf(w[i]);
    smem[threadIdx.x] = acc;
    __syncthreads();
    for (int o = 128; o > 0; o >>= 1) {
        if ((int)threadIdx.x < o) smem[threadIdx.x] += smem[threadIdx.x + o];
        __syncthreads();
    }
    if (threadIdx.x == 0) partial[blockIdx.x] = smem[0];
}

__global__ void absmean_final_kernel(const float* __restrict__ partial, int nparts,
                                     float* __restrict__ out2, float inv_n) {
    __shared__ float smem[256];
    float acc = 0.0f;
    for (int i = threadIdx.x; i < nparts; i += 256) acc += partial[i];
    smem[threadIdx.x] = acc;
    __syncthreads();
    for (int o = 128; o > 0; o >>= 1) {
        if ((int)threadIdx.x < o) smem[threadIdx.x] += smem[threadIdx.x + o];
        __syncthreads();
    }
    if (threadIdx.x == 0) {
        float mean  = smem[0] * inv_n;
        float clipd = fmaxf(mean, EPSQ);
        out2[0] = 1.0f / clipd;
        out2[1] = clipd;
    }
}


__global__ void wquant_kernel(const float* __restrict__ w, int8_t* __restrict__ q,
                              const float* __restrict__ sc, size_t n) {
    const float s = sc[0];
    for (size_t i = (size_t)blockIdx.x * 256 + threadIdx.x; i < n;
         i += (size_t)gridDim.x * 256) {
        float v = rintf(w[i] * s);
        v = fminf(fmaxf(v, -1.0f), 1.0f);
        q[i] = (int8_t)(int)v;
    }
}


__global__ void aquant_kernel(const float* __restrict__ x, int8_t* __restrict__ q,
                              float* __restrict__ srecip, int K) {
    __shared__ float smem[256];
    const size_t row = blockIdx.x;
    const float* xr = x + row * (size_t)K;
    float amax = 0.0f;
    for (int i = threadIdx.x; i < K; i += 256) amax = fmaxf(amax, fabsf(xr[i]));
    smem[threadIdx.x] = amax;
    __syncthreads();
    for (int o = 128; o > 0; o >>= 1) {
        if ((int)threadIdx.x < o)
            smem[threadIdx.x] = fmaxf(smem[threadIdx.x], smem[threadIdx.x + o]);
        __syncthreads();
    }
    const float clipd = fmaxf(smem[0], EPSQ);
    const float scale = 127.0f / clipd;
    if (threadIdx.x == 0) srecip[row] = clipd / 127.0f;
    int8_t* qr = q + row * (size_t)K;
    for (int i = threadIdx.x; i < K; i += 256) {
        float v = rintf(xr[i] * scale);
        v = fminf(fmaxf(v, -128.0f), 127.0f);
        qr[i] = (int8_t)(int)v;
    }
}

__device__ __forceinline__ v8i load_a_iu8(const int8_t* __restrict__ base, int K,
                                          int m0, int k0, int lane) {
    const int row = lane & 15;
    const int hi  = lane >> 4;
    const int8_t* p = base + (size_t)(m0 + row) * K + k0 + hi * 16;
    v8i a;
#pragma unroll
    for (int i = 0; i < 4; ++i) {
        uint2 d = *(const uint2*)(p + (i & 1) * 8 + (i >> 1) * 32);
        a[2 * i]     = (int)d.x;
        a[2 * i + 1] = (int)d.y;
    }
    return a;
}

__device__ __forceinline__ v8i load_b_iu8(const int8_t* __restrict__ base, int K,
                                          int n0, int k0, int lane) {
    const int col = lane & 15;
    const int hi  = lane >> 4;
    const int8_t* p = base + (size_t)(n0 + col) * K + k0 + hi * 16;
    uint4 lo = *(const uint4*)(p);
    uint4 up = *(const uint4*)(p + 32);
    v8i b;
    b[0] = (int)lo.x; b[1] = (int)lo.y; b[2] = (int)lo.z; b[3] = (int)lo.w;
    b[4] = (int)up.x; b[5] = (int)up.y; b[6] = (int)up.z; b[7] = (int)up.w;
    return b;
}

__global__ void __launch_bounds__(256)
gate_gemm_kernel(const int8_t* __restrict__ xq, const int8_t* __restrict__ w1q,
                 const int8_t* __restrict__ w3q, const float* __restrict__ xsr,
                 const float* __restrict__ sc,
                 float* __restrict__ h, int M, int K, int N) {
    const int wave = (int)((blockIdx.x * blockDim.x + threadIdx.x) >> 5);
    const int lane = (int)(threadIdx.x & 31);
    const int stripsN = N / (16 * NSUB);
    const int mt = wave / stripsN;
    const int ns = wave % stripsN;
    if (mt >= (M >> 4)) return;
    const int m0 = mt << 4;
    const int n0 = ns * (16 * NSUB);

    v8i acc1[NSUB];
    v8i acc3[NSUB];
#pragma unroll
    for (int j = 0; j < NSUB; ++j) { acc1[j] = (v8i){}; acc3[j] = (v8i){}; }

    for (int k0 = 0; k0 < K; k0 += 64) {
        if (k0 + 64 < K)
            __builtin_prefetch(xq + (size_t)(m0 + (lane & 15)) * K + k0 + 64, 0, 1);
        const v8i a = load_a_iu8(xq, K, m0, k0, lane);
#pragma unroll
        for (int j = 0; j < NSUB; ++j) {
            v8i b1 = load_b_iu8(w1q, K, n0 + 16 * j, k0, lane);
            acc1[j] = __builtin_amdgcn_wmma_i32_16x16x64_iu8(true, a, true, b1,
                                                             acc1[j], false, false);
            v8i b3 = load_b_iu8(w3q, K, n0 + 16 * j, k0, lane);
            acc3[j] = __builtin_amdgcn_wmma_i32_16x16x64_iu8(true, a, true, b3,
                                                             acc3[j], false, false);
        }
    }

    const float r1 = sc[1];
    const float r3 = sc[5];
    const int hi  = lane >> 4;
    const int col = lane & 15;
#pragma unroll
    for (int g = 0; g < 8; ++g) {
        const int m = m0 + g + hi * 8;
        const float s  = xsr[m];
        const float s1 = s * r1;
        const float s3 = s * r3;
        float* hrow = h + (size_t)m * N + n0 + col;
#pragma unroll
        for (int j = 0; j < NSUB; ++j) {
            const float a1 = (float)acc1[j][g] * s1;
            const float a3 = (float)acc3[j][g] * s3;
            hrow[16 * j] = (a1 / (1.0f + __expf(-a1))) * a3;
        }
    }
}


__global__ void __launch_bounds__(256)
out_gemm_kernel(const int8_t* __restrict__ hq, const int8_t* __restrict__ w2q,
                const float* __restrict__ hsr,
                const float* __restrict__ sc,
                float* __restrict__ out, int M, int K, int N) {
    const int wave = (int)((blockIdx.x * blockDim.x + threadIdx.x) >> 5);
    const int lane = (int)(threadIdx.x & 31);
    const int stripsN = N / (16 * NSUB);
    const int mt = wave / stripsN;
    const int ns = wave % stripsN;
    if (mt >= (M >> 4)) return;
    const int m0 = mt << 4;
    const int n0 = ns * (16 * NSUB);

    v8i acc[NSUB];
#pragma unroll
    for (int j = 0; j < NSUB; ++j) acc[j] = (v8i){};

    for (int k0 = 0; k0 < K; k0 += 64) {
        if (k0 + 64 < K)
            __builtin_prefetch(hq + (size_t)(m0 + (lane & 15)) * K + k0 + 64, 0, 1);
        const v8i a = load_a_iu8(hq, K, m0, k0, lane);
#pragma unroll
        for (int j = 0; j < NSUB; ++j) {
            v8i b = load_b_iu8(w2q, K, n0 + 16 * j, k0, lane);
            acc[j] = __builtin_amdgcn_wmma_i32_16x16x64_iu8(true, a, true, b,
                                                            acc[j], false, false);
        }
    }

    const float r2 = sc[3];
    const int hi  = lane >> 4;
    const int col = lane & 15;
#pragma unroll
    for (int g = 0; g < 8; ++g) {
        const int m = m0 + g + hi * 8;
        const float s = hsr[m] * r2;
        float* orow = out + (size_t)m * N + n0 + col;
#pragma unroll
        for (int j = 0; j < NSUB; ++j) orow[16 * j] = (float)acc[j][g] * s;
    }
}


extern "C" void kernel_launch(void* const* d_in, const int* in_sizes, int n_in,
                              void* d_out, int out_size, void* d_ws, size_t ws_size,
                              hipStream_t stream) {
    const float* x  = (const float*)d_in[0];
    const float* w1 = (const float*)d_in[1];
    const float* w2 = (const float*)d_in[2];
    const float* w3 = (const float*)d_in[3];

    const int D = 2048;
    const int H = 5632;
    const int M = in_sizes[0] / D;
    const size_t nW1 = (size_t)in_sizes[1];
    const size_t nW2 = (size_t)in_sizes[2];
    const size_t nW3 = (size_t)in_sizes[3];

    char* ws = (char*)d_ws;
    size_t off = 0;
    auto carve = [&](size_t bytes) -> void* {
        off = (off + 255) & ~(size_t)255;
        void* p = (void*)(ws + off);
        off += bytes;
        return p;
    };
    float*  partials = (float*)carve(1024 * sizeof(float));
    float*  scales   = (float*)carve(8 * sizeof(float));
    int8_t* w1q = (int8_t*)carve(nW1);
    int8_t* w2q = (int8_t*)carve(nW2);
    int8_t* w3q = (int8_t*)carve(nW3);
    int8_t* xq  = (int8_t*)carve((size_t)M * D);
    float*  xsr = (float*)carve((size_t)M * sizeof(float));
    float*  hbuf = (float*)carve((size_t)M * H * sizeof(float));
    int8_t* hq  = (int8_t*)carve((size_t)M * H);
    float*  hsr = (float*)carve((size_t)M * sizeof(float));
    (void)ws_size;

    const int RB = 512;

    absmean_partial_kernel<<<RB, 256, 0, stream>>>(w1, partials, nW1);
    absmean_final_kernel<<<1, 256, 0, stream>>>(partials, RB, scales + 0,
                                                1.0f / (float)nW1);
    absmean_partial_kernel<<<RB, 256, 0, stream>>>(w2, partials, nW2);
    absmean_final_kernel<<<1, 256, 0, stream>>>(partials, RB, scales + 2,
                                                1.0f / (float)nW2);
    absmean_partial_kernel<<<RB, 256, 0, stream>>>(w3, partials, nW3);
    absmean_final_kernel<<<1, 256, 0, stream>>>(partials, RB, scales + 4,
                                                1.0f / (float)nW3);

    const int QB = 2048;
    wquant_kernel<<<QB, 256, 0, stream>>>(w1, w1q, scales + 0, nW1);
    wquant_kernel<<<QB, 256, 0, stream>>>(w2, w2q, scales + 2, nW2);
    wquant_kernel<<<QB, 256, 0, stream>>>(w3, w3q, scales + 4, nW3);

    aquant_kernel<<<M, 256, 0, stream>>>(x, xq, xsr, D);

    {
        const long waves = (long)(M >> 4) * (H / (16 * NSUB));
        const int blocks = (int)((waves + 7) / 8);
        gate_gemm_kernel<<<blocks, 256, 0, stream>>>(xq, w1q, w3q, xsr, scales,
                                                     hbuf, M, D, H);
    }

    aquant_kernel<<<M, 256, 0, stream>>>(hbuf, hq, hsr, H);

    {
        const long waves = (long)(M >> 4) * (D / (16 * NSUB));
        const int blocks = (int)((waves + 7) / 8);
        out_gemm_kernel<<<blocks, 256, 0, stream>>>(hq, w2q, hsr, scales,
                                                    (float*)d_out, M, H, D);
    }
    (void)n_in; (void)out_size;
}
